// TASA_9517647527993
// MI455X (gfx1250) — matrix operand order repaired and checked by reading, not compiled, not run on hardware
//
#include <hip/hip_runtime.h>
#include <math.h>

#define VOCAB 10000
#define EMB   64
#define HID   128
#define SEQ   100
#define BATCH 256
#define G4    512           // 4*HID
#define MTB   (SEQ*BATCH)   // 25600
#define NT    5             // N-tiles per block in fused logits kernel (625 = 125*5)
#define NCOLS (NT*16)       // 80 columns per block
#define BSTRIDE 136         // padded LDS row stride (bf16 elems): 272B -> 4-bank skew

typedef __attribute__((ext_vector_type(16))) __bf16 v16bf;
typedef __attribute__((ext_vector_type(8)))  float  v8f;
typedef __attribute__((ext_vector_type(4)))  unsigned int u32x4;
typedef __attribute__((ext_vector_type(8)))  int i32x8;
typedef __attribute__((ext_vector_type(4)))  int i32x4;

union Frag32B { v16bf bf; float4 f4[2]; };

// ---- WMMA helpers (bf16 x bf16 -> f32, 16x16x32) --------------------------

__device__ __forceinline__ v8f wmma_bf16(v16bf a, v16bf b, v8f c) {
    return __builtin_amdgcn_wmma_f32_16x16x32_bf16(
        /*neg_a=*/false, a, /*neg_b=*/false, b,
        /*c_mod=*/(short)0, c, /*reuse_a=*/false, /*reuse_b=*/false);
}

// A fragment: 16x32 bf16 tile, row M = (row), from row-major [M][ldK].
// Lanes 0-15 hold K {0..7,16..23}, lanes 16-31 hold K {8..15,24..31}.
__device__ __forceinline__ v16bf load_a_frag(const __bf16* base, int row,
                                             int ldK, int kk, int lane) {
    const int hi = lane >> 4;
    const __bf16* p = base + row * ldK + kk + hi * 8;
    Frag32B f;
    f.f4[0] = *(const float4*)(p);
    f.f4[1] = *(const float4*)(p + 16);
    return f.bf;
}

// Same A fragment, from an LDS tile with padded row stride.
__device__ __forceinline__ v16bf load_a_frag_lds(const __bf16* lds, int row,
                                                 int kk, int lane) {
    const int hi = lane >> 4;
    const __bf16* p = lds + row * BSTRIDE + kk + hi * 8;
    Frag32B f;
    f.f4[0] = *(const float4*)(p);
    f.f4[1] = *(const float4*)(p + 16);
    return f.bf;
}

// B fragment: 32x16 bf16 tile, column N = ncol (row of row-major [N][ldK]).
// Lanes 0-15 hold K {0..7,16..23}, lanes 16-31 hold K {8..15,24..31}.
__device__ __forceinline__ v16bf load_b_frag(const __bf16* base, int ncol,
                                             int ldK, int kk, int lane) {
    const int hi = lane >> 4;
    const __bf16* p = base + ncol * ldK + kk + hi * 8;
    Frag32B f;
    f.f4[0] = *(const float4*)(p);
    f.f4[1] = *(const float4*)(p + 16);
    return f.bf;
}

// Same B fragment, sourced from an LDS tile with padded row stride.
__device__ __forceinline__ v16bf load_b_frag_lds(const __bf16* lds, int nloc,
                                                 int kk, int lane) {
    const int hi = lane >> 4;
    const __bf16* p = lds + nloc * BSTRIDE + kk + hi * 8;
    Frag32B f;
    f.f4[0] = *(const float4*)(p);
    f.f4[1] = *(const float4*)(p + 16);
    return f.bf;
}

__device__ __forceinline__ float sigm(float x) {
    return 1.0f / (1.0f + __expf(-x));
}

// ---- Tensor Data Mover: stage a [rows x 128] bf16 row-major tile into LDS
// with the padded BSTRIDE layout (pad 16B after every 256B row).
// D# built per CDNA5 ISA ch.8: 2-D tile, data_size=2B, pad_interval=64 DWORDs
// (code 5), pad_amount=4 DWORDs (code 3) -> LDS row stride 272B = BSTRIDE*2.
// Issued by one wave; completion via TENSORcnt, then block barrier.

__device__ __forceinline__ void tdm_stage_tile(const __bf16* gsrc,
                                               __bf16* lds_dst, int rows) {
    unsigned long long ga = (unsigned long long)(uintptr_t)gsrc;
    unsigned int la = (unsigned int)(uintptr_t)lds_dst; // LDS offset = low 32b
    u32x4 g0;
    g0.x = 1u;                                   // count=1, user mode
    g0.y = la;                                   // lds_addr
    g0.z = (unsigned int)ga;                     // global_addr[31:0]
    g0.w = (unsigned int)((ga >> 32) & 0x01FFFFFFu) | (2u << 30); // addr[56:32] | type=2
    i32x8 g1;
    g1[0] = (int)((1u << 16) | (1u << 20) | (5u << 22) | (3u << 25));
            // data_size=2B | pad_enable | pad_interval=64DW | pad_amount=4DW
    g1[1] = (int)(128u << 16);                   // tensor_dim0 = 128 (bits 79:48 lo)
    g1[2] = (int)(((unsigned)rows & 0xFFFFu) << 16); // dim0 hi=0 | tensor_dim1 lo
    g1[3] = (int)(128u << 16);                   // tensor_dim1 hi=0 | tile_dim0 = 128
    g1[4] = (int)((unsigned)rows & 0xFFFFu);     // tile_dim1 = rows | tile_dim2 = 0
    g1[5] = 128;                                 // tensor_dim0_stride = 128 elems
    g1[6] = 0;
    g1[7] = 0;
    i32x4 z4 = {0, 0, 0, 0};
#if defined(__clang_major__) && (__clang_major__ >= 23)
    i32x8 z8 = {0, 0, 0, 0, 0, 0, 0, 0};
    __builtin_amdgcn_tensor_load_to_lds(g0, g1, z4, z4, z8, 0);
#else
    __builtin_amdgcn_tensor_load_to_lds(g0, g1, z4, z4, 0);
#endif
    __builtin_amdgcn_s_wait_tensorcnt((short)0);
}

// ---- Stage 0: deltas = sigmoid(taus @ W_infl^T + b_infl) ------------------

__global__ void k_deltas(const float* __restrict__ taus,
                         const float* __restrict__ Winfl,
                         const float* __restrict__ binfl,
                         float* __restrict__ deltas) {
    int idx = blockIdx.x * blockDim.x + threadIdx.x;
    if (idx >= BATCH * SEQ) return;
    int b = idx / SEQ, t = idx % SEQ;
    float acc = binfl[t];
    const float* tr = taus + b * SEQ;
    const float* wr = Winfl + t * SEQ;
    #pragma unroll 4
    for (int k = 0; k < SEQ; ++k) acc += tr[k] * wr[k];
    deltas[idx] = sigm(acc);
}

// ---- Stage 1: x[t][b][e] = emb[sessions[b][t]][e] * deltas[b][t] (bf16) ---

__global__ void k_build_x(const int* __restrict__ sessions,
                          const float* __restrict__ emb,
                          const float* __restrict__ deltas,
                          __bf16* __restrict__ x) {
    int idx = blockIdx.x * blockDim.x + threadIdx.x;
    if (idx >= SEQ * BATCH * EMB) return;
    int e  = idx % EMB;
    int tb = idx / EMB;
    int b  = tb % BATCH;
    int t  = tb / BATCH;
    int s  = sessions[b * SEQ + t];
    float v = emb[s * EMB + e] * deltas[b * SEQ + t];
    x[idx] = (__bf16)v;
}

// ---- f32 -> bf16 conversion (weights) -------------------------------------

__global__ void k_cvt_bf16(const float* __restrict__ in,
                           __bf16* __restrict__ out, int n) {
    int i = blockIdx.x * blockDim.x + threadIdx.x;
    if (i < n) out[i] = (__bf16)in[i];
}

// ---- zero LSTM state ------------------------------------------------------

__global__ void k_init_state(__bf16* __restrict__ h, float* __restrict__ c) {
    int i = blockIdx.x * blockDim.x + threadIdx.x;
    if (i < BATCH * HID) { h[i] = (__bf16)0.0f; c[i] = 0.0f; }
}

// ---- Input projection: zx[m][g] = x[m][:] . Wih[g][:] + bih[g]+bhh[g] -----
// M = 25600, N = 512, K = 64. grid (1600, 8), block = 32 (1 wave).

__global__ void k_zx_gemm(const __bf16* __restrict__ x,
                          const __bf16* __restrict__ Wih,
                          const float* __restrict__ bih,
                          const float* __restrict__ bhh,
                          float* __restrict__ zx) {
    const int lane = threadIdx.x;
    const int lo = lane & 15, hi = lane >> 4;
    const int m0 = blockIdx.x * 16;
    v16bf a0 = load_a_frag(x, m0 + lo, EMB, 0,  lane);
    v16bf a1 = load_a_frag(x, m0 + lo, EMB, 32, lane);
    #pragma unroll
    for (int i = 0; i < 4; ++i) {
        const int ncol = (blockIdx.y * 4 + i) * 16 + lo;
        v16bf b0 = load_b_frag(Wih, ncol, EMB, 0,  lane);
        v16bf b1 = load_b_frag(Wih, ncol, EMB, 32, lane);
        v8f acc = {};
        acc = wmma_bf16(a0, b0, acc);
        acc = wmma_bf16(a1, b1, acc);
        const float bias = bih[ncol] + bhh[ncol];
        #pragma unroll
        for (int r = 0; r < 8; ++r) {
            int m = m0 + hi * 8 + r;
            zx[m * G4 + ncol] = acc[r] + bias;
        }
    }
}

// ---- Whole-sequence LSTM: 100 steps in ONE kernel -------------------------
// The recurrence is independent per batch row, so a block owning batch rows
// 16m..16m+15 never needs data from other blocks: the full time loop runs
// inside the kernel with only block barriers. h (bf16) and c (f32) stay in
// LDS for all 100 steps; Whh (512x128 bf16, 139 KB) is staged once into the
// 320 KB WGP LDS by the Tensor Data Mover with the skewed-row padding.
// grid = 16 blocks x 512 threads (16 waves, wave w -> N-tiles 2w, 2w+1).

__global__ void __launch_bounds__(512)
k_lstm_seq(const float* __restrict__ zx,
           const __bf16* __restrict__ Whh,
           __bf16* __restrict__ h_g,
           float* __restrict__ c_g,
           __bf16* __restrict__ hs,
           int write_hs) {
    __shared__ float  zsh[16 * G4];          // 32 KB  gate pre-activations
    __shared__ __bf16 hsh[16 * BSTRIDE];     // 4.25 KB hidden state (padded)
    __shared__ float  csh[16 * HID];         // 8 KB   cell state
    __shared__ __bf16 Wl[G4 * BSTRIDE];      // 139 KB staged Whh (padded)

    const int tid  = threadIdx.x;
    const int lane = tid & 31;
    const int w    = tid >> 5;
    const int lo = lane & 15, hi = lane >> 4;
    const int m0 = blockIdx.x * 16;

    if (tid < 32) tdm_stage_tile(Whh, Wl, G4);   // wave 0: TDM + tensorcnt wait

    for (int i = tid; i < 16 * HID; i += 512) {  // load h, c into LDS
        int r = i >> 7, j = i & 127;
        hsh[r * BSTRIDE + j] = h_g[(m0 + r) * HID + j];
        csh[i] = c_g[(m0 + r) * HID + j];
    }
    __syncthreads();

    for (int t = 0; t < SEQ; ++t) {
        // recurrent GEMM: z = zx[t] + h @ Whh^T
        v16bf a[4];
        #pragma unroll
        for (int kt = 0; kt < 4; ++kt)
            a[kt] = load_a_frag_lds(hsh, lo, kt * 32, lane);
        #pragma unroll
        for (int i = 0; i < 2; ++i) {
            const int ncol = (w * 2 + i) * 16 + lo;
            v8f acc;
            #pragma unroll
            for (int r = 0; r < 8; ++r)
                acc[r] = zx[(t * BATCH + m0 + hi * 8 + r) * G4 + ncol];
            #pragma unroll
            for (int kt = 0; kt < 4; ++kt) {
                v16bf bf = load_b_frag_lds(Wl, ncol, kt * 32, lane);
                acc = wmma_bf16(a[kt], bf, acc);
            }
            #pragma unroll
            for (int r = 0; r < 8; ++r)
                zsh[(hi * 8 + r) * G4 + ncol] = acc[r];
        }
        __syncthreads();

        // gates: 16 rows x 128 hidden = 2048 elems, 4 per thread
        #pragma unroll
        for (int e = 0; e < 4; ++e) {
            int flat = e * 512 + tid;
            int bl = flat >> 7;
            int j  = flat & 127;
            float zi = zsh[bl * G4 + j];
            float zf = zsh[bl * G4 + 128 + j];
            float zg = zsh[bl * G4 + 256 + j];
            float zo = zsh[bl * G4 + 384 + j];
            float cn = sigm(zf) * csh[flat] + sigm(zi) * tanhf(zg);
            float hn = sigm(zo) * tanhf(cn);
            csh[flat] = cn;
            __bf16 hb = (__bf16)hn;
            hsh[bl * BSTRIDE + j] = hb;
            if (write_hs) hs[(t * BATCH + m0 + bl) * HID + j] = hb;
        }
        __syncthreads();
    }

    for (int i = tid; i < 16 * HID; i += 512) {  // write back final h, c
        int r = i >> 7, j = i & 127;
        h_g[(m0 + r) * HID + j] = hsh[r * BSTRIDE + j];
        c_g[(m0 + r) * HID + j] = csh[i];
    }
}

// ---- Fused logits GEMM + softmax over batch axis --------------------------
// logits[b][t][v] = hs[t][b][:] . Wout[v][:]   (b_out cancels under softmax
// over the batch axis). Block (vg, t) owns all 256 batch rows of time t and
// 80 vocab columns; logits never leave registers; one pass writes d_out.

__global__ void __launch_bounds__(512)
k_logits_softmax(const __bf16* __restrict__ hs,
                 const __bf16* __restrict__ Wout,
                 float* __restrict__ out) {
    __shared__ __bf16 Bl[NCOLS * BSTRIDE];        // staged W_out tile (~21.8 KB)
    __shared__ float  red[16][2][NCOLS];
    __shared__ float  colmax[NCOLS];
    __shared__ float  colsum[NCOLS];

    const int tid  = threadIdx.x;
    const int lane = tid & 31;
    const int w    = tid >> 5;
    const int lo = lane & 15, hi = lane >> 4;
    const int t  = blockIdx.y;
    const int v0 = blockIdx.x * NCOLS;

    if (tid < 32) tdm_stage_tile(Wout + (size_t)v0 * HID, Bl, NCOLS);

    const int m0 = t * BATCH + w * 16;
    v16bf a[4];
    #pragma unroll
    for (int kt = 0; kt < 4; ++kt)
        a[kt] = load_a_frag(hs, m0 + lo, HID, kt * 32, lane);

    __syncthreads();

    v8f acc[NT];
    #pragma unroll
    for (int i = 0; i < NT; ++i) {
        v8f c = {};
        #pragma unroll
        for (int kt = 0; kt < 4; ++kt) {
            v16bf bf = load_b_frag_lds(Bl, i * 16 + lo, kt * 32, lane);
            c = wmma_bf16(a[kt], bf, c);
        }
        acc[i] = c;
    }

    // 1) column max over b
    #pragma unroll
    for (int i = 0; i < NT; ++i) {
        float pm = acc[i][0];
        #pragma unroll
        for (int r = 1; r < 8; ++r) pm = fmaxf(pm, acc[i][r]);
        red[w][hi][i * 16 + lo] = pm;
    }
    __syncthreads();
    if (tid < NCOLS) {
        float m = -3.402823466e38f;
        #pragma unroll
        for (int ww = 0; ww < 16; ++ww) {
            m = fmaxf(m, red[ww][0][tid]);
            m = fmaxf(m, red[ww][1][tid]);
        }
        colmax[tid] = m;
    }
    __syncthreads();
    // 2) exponentiate in-register, column sum
    #pragma unroll
    for (int i = 0; i < NT; ++i) {
        float cm = colmax[i * 16 + lo];
        float ps = 0.0f;
        #pragma unroll
        for (int r = 0; r < 8; ++r) {
            float e = __expf(acc[i][r] - cm);
            acc[i][r] = e;
            ps += e;
        }
        red[w][hi][i * 16 + lo] = ps;
    }
    __syncthreads();
    if (tid < NCOLS) {
        float s = 0.0f;
        #pragma unroll
        for (int ww = 0; ww < 16; ++ww)
            s += red[ww][0][tid] + red[ww][1][tid];
        colsum[tid] = s;
    }
    __syncthreads();
    // 3) scaled write: out[b][t][v]
    #pragma unroll
    for (int i = 0; i < NT; ++i) {
        float inv = 1.0f / colsum[i * 16 + lo];
        int v = v0 + i * 16 + lo;
        #pragma unroll
        for (int r = 0; r < 8; ++r) {
            int b = w * 16 + hi * 8 + r;
            out[((size_t)b * SEQ + t) * VOCAB + v] = acc[i][r] * inv;
        }
    }
}

// ---- Host orchestration ---------------------------------------------------

static inline size_t align256(size_t x) { return (x + 255) & ~(size_t)255; }

extern "C" void kernel_launch(void* const* d_in, const int* in_sizes, int n_in,
                              void* d_out, int out_size, void* d_ws, size_t ws_size,
                              hipStream_t stream) {
    const int*   sessions = (const int*)  d_in[0];
    const float* taus     = (const float*)d_in[1];
    const float* emb      = (const float*)d_in[2];
    const float* W_infl   = (const float*)d_in[3];
    const float* b_infl   = (const float*)d_in[4];
    const float* enc_Wih  = (const float*)d_in[5];
    const float* enc_Whh  = (const float*)d_in[6];
    const float* enc_bih  = (const float*)d_in[7];
    const float* enc_bhh  = (const float*)d_in[8];
    const float* dec_Wih  = (const float*)d_in[9];
    const float* dec_Whh  = (const float*)d_in[10];
    const float* dec_bih  = (const float*)d_in[11];
    const float* dec_bhh  = (const float*)d_in[12];
    const float* W_out    = (const float*)d_in[13];
    float* out = (float*)d_out;

    char* ws = (char*)d_ws;
    size_t off = 0;
    float*  deltas   = (float*)(ws + off); off += align256(BATCH * SEQ * 4);
    __bf16* x_bf     = (__bf16*)(ws + off); off += align256((size_t)SEQ * BATCH * EMB * 2);
    __bf16* WihE     = (__bf16*)(ws + off); off += align256(G4 * EMB * 2);
    __bf16* WhhE     = (__bf16*)(ws + off); off += align256(G4 * HID * 2);
    __bf16* WihD     = (__bf16*)(ws + off); off += align256(G4 * EMB * 2);
    __bf16* WhhD     = (__bf16*)(ws + off); off += align256(G4 * HID * 2);
    __bf16* WoutB    = (__bf16*)(ws + off); off += align256((size_t)VOCAB * HID * 2);
    float*  zx       = (float*)(ws + off); off += align256((size_t)MTB * G4 * 4);
    __bf16* h_st     = (__bf16*)(ws + off); off += align256(BATCH * HID * 2);
    float*  c_st     = (float*)(ws + off); off += align256(BATCH * HID * 4);
    __bf16* hs_dec   = (__bf16*)(ws + off); off += align256((size_t)MTB * HID * 2);
    if (off > ws_size) return;

    k_deltas<<<(BATCH * SEQ + 255) / 256, 256, 0, stream>>>(taus, W_infl, b_infl, deltas);
    k_build_x<<<(SEQ * BATCH * EMB + 255) / 256, 256, 0, stream>>>(sessions, emb, deltas, x_bf);
    k_cvt_bf16<<<(G4 * EMB + 255) / 256, 256, 0, stream>>>(enc_Wih, WihE, G4 * EMB);
    k_cvt_bf16<<<(G4 * HID + 255) / 256, 256, 0, stream>>>(enc_Whh, WhhE, G4 * HID);
    k_cvt_bf16<<<(G4 * EMB + 255) / 256, 256, 0, stream>>>(dec_Wih, WihD, G4 * EMB);
    k_cvt_bf16<<<(G4 * HID + 255) / 256, 256, 0, stream>>>(dec_Whh, WhhD, G4 * HID);
    k_cvt_bf16<<<(VOCAB * HID + 255) / 256, 256, 0, stream>>>(W_out, WoutB, VOCAB * HID);
    k_init_state<<<(BATCH * HID + 255) / 256, 256, 0, stream>>>(h_st, c_st);

    // encoder: input projection + whole 100-step sequence in one launch
    k_zx_gemm<<<dim3(MTB / 16, G4 / 64), 32, 0, stream>>>(x_bf, WihE, enc_bih, enc_bhh, zx);
    k_lstm_seq<<<BATCH / 16, 512, 0, stream>>>(zx, WhhE, h_st, c_st, hs_dec, 0);

    // decoder (reuses zx buffer; state carries over from encoder)
    k_zx_gemm<<<dim3(MTB / 16, G4 / 64), 32, 0, stream>>>(x_bf, WihD, dec_bih, dec_bhh, zx);
    k_lstm_seq<<<BATCH / 16, 512, 0, stream>>>(zx, WhhD, h_st, c_st, hs_dec, 1);

    // fused logits GEMM + batch-axis softmax, single pass over d_out
    k_logits_softmax<<<dim3(VOCAB / 16 / NT, SEQ), 512, 0, stream>>>(hs_dec, WoutB, out);
}
